// Attention_1726576853155
// MI455X (gfx1250) — matrix operand order repaired and checked by reading, not compiled, not run on hardware
//
#include <hip/hip_runtime.h>

// ---------------------------------------------------------------------------
// MI455X (gfx1250) attention: bf16 WMMA (v_wmma_f32_16x16x32_bf16) everywhere,
// f32 accumulate.  Attention streams K/V tiles into LDS with async
// global->LDS copies (ASYNCcnt), double-buffered, flash-style softmax in the
// exp2 domain, V pre-transposed so P*V B-fragments are contiguous LDS loads.
// ---------------------------------------------------------------------------

typedef __attribute__((ext_vector_type(16))) __bf16 bf16x16;
typedef __attribute__((ext_vector_type(8)))  float  f32x8;
typedef __attribute__((ext_vector_type(4)))  int    i32x4;
typedef unsigned short u16;
typedef unsigned int   u32;

#define NB      2
#define SEQ     2048
#define DMODEL  1024
#define NH      16
#define HD      64
#define KT      32              // key-tile size in attention

#if __has_builtin(__builtin_amdgcn_global_load_async_to_lds_b128)
#define HAVE_ASYNC 1
#else
#define HAVE_ASYNC 0
#endif

// address-space-qualified vector pointers for the async builtin
typedef __attribute__((address_space(1))) i32x4 gv4i;   // global
typedef __attribute__((address_space(3))) i32x4 lv4i;   // LDS

__device__ __forceinline__ u16 f32_to_bf16(float f) {
  u32 u = __float_as_uint(f);
  u32 r = u + 0x7FFFu + ((u >> 16) & 1u);   // round-to-nearest-even
  return (u16)(r >> 16);
}

// Two 16B chunks -> one 16x bf16 fragment (32 bytes / 8 VGPRs per lane).
__device__ __forceinline__ bf16x16 load_frag2(const u16* p0, const u16* p1) {
  union { bf16x16 v; uint4 q[2]; } u;
  u.q[0] = *reinterpret_cast<const uint4*>(p0);
  u.q[1] = *reinterpret_cast<const uint4*>(p1);
  return u.v;
}

__device__ __forceinline__ f32x8 wmma_bf16(bf16x16 a, bf16x16 b, f32x8 c) {
  // (neg_a, A, neg_b, B, c_mod, C, reuse_a, reuse_b)
  return __builtin_amdgcn_wmma_f32_16x16x32_bf16(false, a, false, b,
                                                 (short)0, c, false, false);
}

__device__ __forceinline__ f32x8 zero8() {
  f32x8 v;
#pragma unroll
  for (int i = 0; i < 8; ++i) v[i] = 0.0f;
  return v;
}

// 16B global -> LDS copy.  Async (no VGPR round-trip, ASYNCcnt) when the
// toolchain declares the gfx1250 builtin; synchronous bounce otherwise.
__device__ __forceinline__ void cp16_g2l(const u16* gsrc, u16* ldst) {
#if HAVE_ASYNC
  __builtin_amdgcn_global_load_async_to_lds_b128(
      (gv4i*)gsrc, (lv4i*)ldst, 0, 0);
#else
  *reinterpret_cast<uint4*>(ldst) = *reinterpret_cast<const uint4*>(gsrc);
#endif
}

__device__ __forceinline__ void wait_async() {
#if HAVE_ASYNC
#if __has_builtin(__builtin_amdgcn_s_wait_asynccnt)
  __builtin_amdgcn_s_wait_asynccnt(0);
#else
  asm volatile("s_wait_asynccnt 0" ::: "memory");
#endif
#endif
}

// ---------------------------------------------------------------------------
// Kernel 1: fp32 -> bf16 conversion
// ---------------------------------------------------------------------------
__global__ void cvt_f32_bf16_kernel(const float* __restrict__ src,
                                    u16* __restrict__ dst, int n) {
  int i = blockIdx.x * blockDim.x + threadIdx.x;
  if (i < n) dst[i] = f32_to_bf16(src[i]);
}

// ---------------------------------------------------------------------------
// Kernel 2: fused QKV projection.  X[4096,1024] @ W.T (W row-major [N,K]).
// blockIdx.y: 0=Q, 1=K, 2=V.  Q/K -> [B,H,S,hd]; V -> transposed [B,H,hd,S].
// Block = 256 threads = 8 waves; wave computes a 16(M) x 64(N) tile.
// ---------------------------------------------------------------------------
__global__ __launch_bounds__(256)
void qkv_kernel(const u16* __restrict__ X,
                const u16* __restrict__ Wq, const u16* __restrict__ Wk,
                const u16* __restrict__ Wv,
                u16* __restrict__ Q, u16* __restrict__ Kout,
                u16* __restrict__ Vt) {
  const int wave = threadIdx.x >> 5;
  const int lane = threadIdx.x & 31;
  const int mBlocks = (NB * SEQ) / 128;            // 32
  const int mb = blockIdx.x % mBlocks;
  const int nb = blockIdx.x / mBlocks;             // 0..15
  const int which = blockIdx.y;                    // 0=Q 1=K 2=V
  const u16* W = (which == 0) ? Wq : (which == 1) ? Wk : Wv;

  const int m0 = mb * 128 + wave * 16;
  const int n0 = nb * 64;

  const int rowA   = lane & 15;
  const int koffA  = (lane < 16) ? 0 : 8;
  const int colB   = lane & 15;
  const int khalfB = (lane < 16) ? 0 : 8;

  f32x8 acc[4];
#pragma unroll
  for (int j = 0; j < 4; ++j) acc[j] = zero8();

  for (int k = 0; k < DMODEL; k += 32) {
    const u16* ap = X + (size_t)(m0 + rowA) * DMODEL + k + koffA;
    bf16x16 a = load_frag2(ap, ap + 16);
#pragma unroll
    for (int j = 0; j < 4; ++j) {
      const u16* bp = W + (size_t)(n0 + j * 16 + colB) * DMODEL + k + khalfB;
      bf16x16 b = load_frag2(bp, bp + 16);   // two 8-bf16 chunks of W row (table k order)
      acc[j] = wmma_bf16(a, b, acc[j]);
    }
  }

  // C layout: element r of lane L -> row m0 + 8*(L>=16) + r, col n0+j*16+(L&15)
  const int mrow0 = m0 + ((lane >> 4) << 3);
#pragma unroll
  for (int j = 0; j < 4; ++j) {
    const int n = n0 + j * 16 + (lane & 15);
    const int h = n >> 6, d = n & 63;
#pragma unroll
    for (int r = 0; r < 8; ++r) {
      const int m = mrow0 + r;
      const int b_ = m / SEQ, s = m % SEQ;
      const u16 val = f32_to_bf16(acc[j][r]);
      if (which == 2) {
        Vt[(((size_t)b_ * NH + h) * HD + d) * SEQ + s] = val;    // transposed
      } else {
        u16* dst = (which == 0) ? Q : Kout;
        dst[(((size_t)b_ * NH + h) * SEQ + s) * HD + d] = val;
      }
    }
  }
}

// ---------------------------------------------------------------------------
// Kernel 3: causal flash attention.  Per block: one (b,h), 128 queries,
// 8 waves x 16 queries.  K/V tiles (32 keys) staged in LDS via async
// global->LDS copies, double-buffered; softmax state in registers (exp2
// domain); P permuted C-layout -> A-layout through per-wave LDS.
// ---------------------------------------------------------------------------
__global__ __launch_bounds__(256)
void attn_kernel(const u16* __restrict__ Q, const u16* __restrict__ K,
                 const u16* __restrict__ Vt, u16* __restrict__ Ctx) {
  __shared__ u16 ldsK[2][KT][HD];                  // 2 x 4 KB
  __shared__ u16 ldsV[2][HD][KT];                  // 2 x 4 KB
  __shared__ u16 ldsP[8][16 * KT];                 // 8 KB, per-wave P staging

  const int tid  = threadIdx.x;
  const int wave = tid >> 5;
  const int lane = tid & 31;
  const int qb   = blockIdx.x % (SEQ / 128);
  const int bh   = blockIdx.x / (SEQ / 128);       // 0..31
  const int q0   = qb * 128 + wave * 16;

  const u16* Qh = Q  + (size_t)bh * SEQ * HD;
  const u16* Kh = K  + (size_t)bh * SEQ * HD;
  const u16* Vh = Vt + (size_t)bh * HD * SEQ;

  const int rowA   = lane & 15;
  const int koffA  = (lane < 16) ? 0 : 8;
  const int colB   = lane & 15;
  const int khalfB = (lane < 16) ? 0 : 8;

  // cooperative tile stage: K tile 32x64 (8 chunks/row), V tile 64x32 (4/row)
  auto stage = [&](int buf, int kbase) {
    {
      const int row = tid >> 3, off = (tid & 7) << 3;
      cp16_g2l(Kh + (size_t)(kbase + row) * HD + off, &ldsK[buf][row][off]);
    }
    {
      const int row = tid >> 2, off = (tid & 3) << 3;
      cp16_g2l(Vh + (size_t)row * SEQ + kbase + off, &ldsV[buf][row][off]);
    }
  };

  // Q fragments for d-chunks [0,32) and [32,64)
  bf16x16 aQ[2];
  {
    const u16* p = Qh + (size_t)(q0 + rowA) * HD;
    aQ[0] = load_frag2(p + koffA,      p + koffA + 16);
    aQ[1] = load_frag2(p + 32 + koffA, p + 32 + koffA + 16);
  }

  f32x8 o[4];
#pragma unroll
  for (int j = 0; j < 4; ++j) o[j] = zero8();
  float rm[8], rs[8];
#pragma unroll
  for (int r = 0; r < 8; ++r) { rm[r] = -1e30f; rs[r] = 0.0f; }

  // log2-domain scale: softmax(x/hd) == softmax2(x * (1/hd)*log2(e))
  const float c2 = (1.0f / (float)HD) * 1.44269504088896340736f;
  const int qrow0  = q0 + ((lane >> 4) << 3);      // row of accum element r
  const int ntiles = ((q0 + 15) >> 5) + 1;         // this wave's causal limit
  const int ntmax  = qb * 4 + 4;                   // block-wide tile count

  stage(0, 0);

  for (int kt = 0; kt < ntmax; ++kt) {
    const int buf = kt & 1;
    wait_async();
    __syncthreads();                               // tile kt visible to all
    if (kt + 1 < ntmax) stage(buf ^ 1, (kt + 1) * KT);

    if (kt < ntiles) {
      const int kbase = kt * KT;

      // ---- scores S = Q * K^T (16 x 32) from LDS K tile ----
      f32x8 s0 = zero8(), s1 = zero8();
#pragma unroll
      for (int dk = 0; dk < 2; ++dk) {
        const u16* kp0 = &ldsK[buf][colB][dk * 32 + khalfB];
        const u16* kp1 = &ldsK[buf][16 + colB][dk * 32 + khalfB];
        bf16x16 b0 = load_frag2(kp0, kp0 + 16);
        bf16x16 b1 = load_frag2(kp1, kp1 + 16);
        s0 = wmma_bf16(aQ[dk], b0, s0);
        s1 = wmma_bf16(aQ[dk], b1, s1);
      }

      // ---- scale (log2 domain) + causal mask + row max ----
      const int k0i = kbase + (lane & 15);
      const int k1i = k0i + 16;
      float pm[8];
#pragma unroll
      for (int r = 0; r < 8; ++r) {
        const int q = qrow0 + r;
        float v0 = (k0i <= q) ? s0[r] * c2 : -1e30f;
        float v1 = (k1i <= q) ? s1[r] * c2 : -1e30f;
        s0[r] = v0; s1[r] = v1;
        pm[r] = fmaxf(v0, v1);
      }
#pragma unroll
      for (int m = 1; m <= 8; m <<= 1)
#pragma unroll
        for (int r = 0; r < 8; ++r)
          pm[r] = fmaxf(pm[r], __shfl_xor(pm[r], m, 32));

      // ---- online softmax update (exp2) ----
      float alpha[8];
#pragma unroll
      for (int r = 0; r < 8; ++r) {
        const float mn = fmaxf(rm[r], pm[r]);
        alpha[r] = exp2f(rm[r] - mn);
        rm[r] = mn;
      }
      float psum[8];
#pragma unroll
      for (int r = 0; r < 8; ++r) {
        const float e0 = exp2f(s0[r] - rm[r]);
        const float e1 = exp2f(s1[r] - rm[r]);
        s0[r] = e0; s1[r] = e1;
        psum[r] = e0 + e1;
      }
#pragma unroll
      for (int m = 1; m <= 8; m <<= 1)
#pragma unroll
        for (int r = 0; r < 8; ++r)
          psum[r] += __shfl_xor(psum[r], m, 32);
#pragma unroll
      for (int r = 0; r < 8; ++r) rs[r] = rs[r] * alpha[r] + psum[r];
#pragma unroll
      for (int j = 0; j < 4; ++j)
#pragma unroll
        for (int r = 0; r < 8; ++r) o[j][r] *= alpha[r];

      // ---- C-layout -> A-layout permute of P via per-wave LDS ----
      u16* pl = ldsP[wave];
#pragma unroll
      for (int r = 0; r < 8; ++r) {
        const int row = ((lane >> 4) << 3) + r;
        pl[row * KT + (lane & 15)]      = f32_to_bf16(s0[r]);
        pl[row * KT + 16 + (lane & 15)] = f32_to_bf16(s1[r]);
      }
      asm volatile("s_wait_dscnt 0" ::: "memory"); // same-wave LDS RAW

      bf16x16 aP;
      {
        const u16* pr = pl + rowA * KT + koffA;
        union { bf16x16 v; uint4 q[2]; } u;
        u.q[0] = *reinterpret_cast<const uint4*>(pr);
        u.q[1] = *reinterpret_cast<const uint4*>(pr + 16);
        aP = u.v;
      }

      // ---- O += P * V from LDS V tile (transposed: contiguous keys) ----
#pragma unroll
      for (int j = 0; j < 4; ++j) {
        const u16* vp = &ldsV[buf][j * 16 + colB][khalfB];
        bf16x16 bV = load_frag2(vp, vp + 16);
        o[j] = wmma_bf16(aP, bV, o[j]);
      }
    }
  }

  // ---- normalize and store context as [B, S, H*hd] (bf16) ----
  const int b_ = bh / NH, h = bh % NH;
#pragma unroll
  for (int j = 0; j < 4; ++j) {
    const int d = j * 16 + (lane & 15);
#pragma unroll
    for (int r = 0; r < 8; ++r) {
      const int q = qrow0 + r;
      const float val = o[j][r] / rs[r];
      Ctx[((size_t)b_ * SEQ + q) * DMODEL + h * HD + d] = f32_to_bf16(val);
    }
  }
}

// ---------------------------------------------------------------------------
// Kernel 4: output projection  Out = Ctx @ Wo.T + bo  (fp32 out)
// ---------------------------------------------------------------------------
__global__ __launch_bounds__(256)
void out_proj_kernel(const u16* __restrict__ Ctx, const u16* __restrict__ Wo,
                     const float* __restrict__ bo, float* __restrict__ Out) {
  const int wave = threadIdx.x >> 5;
  const int lane = threadIdx.x & 31;
  const int mBlocks = (NB * SEQ) / 128;            // 32
  const int mb = blockIdx.x % mBlocks;
  const int nb = blockIdx.x / mBlocks;

  const int m0 = mb * 128 + wave * 16;
  const int n0 = nb * 64;

  const int rowA   = lane & 15;
  const int koffA  = (lane < 16) ? 0 : 8;
  const int colB   = lane & 15;
  const int khalfB = (lane < 16) ? 0 : 8;

  f32x8 acc[4];
#pragma unroll
  for (int j = 0; j < 4; ++j) acc[j] = zero8();

  for (int k = 0; k < DMODEL; k += 32) {
    const u16* ap = Ctx + (size_t)(m0 + rowA) * DMODEL + k + koffA;
    bf16x16 a = load_frag2(ap, ap + 16);
#pragma unroll
    for (int j = 0; j < 4; ++j) {
      const u16* bp = Wo + (size_t)(n0 + j * 16 + colB) * DMODEL + k + khalfB;
      bf16x16 b = load_frag2(bp, bp + 16);
      acc[j] = wmma_bf16(a, b, acc[j]);
    }
  }

  const int mrow0 = m0 + ((lane >> 4) << 3);
#pragma unroll
  for (int j = 0; j < 4; ++j) {
    const int n = n0 + j * 16 + (lane & 15);
    const float bias = bo[n];
#pragma unroll
    for (int r = 0; r < 8; ++r) {
      const int m = mrow0 + r;
      Out[(size_t)m * DMODEL + n] = acc[j][r] + bias;
    }
  }
}

// ---------------------------------------------------------------------------
// Host entry
// ---------------------------------------------------------------------------
extern "C" void kernel_launch(void* const* d_in, const int* in_sizes, int n_in,
                              void* d_out, int out_size, void* d_ws,
                              size_t ws_size, hipStream_t stream) {
  const float* hs = (const float*)d_in[0];
  const float* Wq = (const float*)d_in[1];
  const float* Wk = (const float*)d_in[2];
  const float* Wv = (const float*)d_in[3];
  const float* Wo = (const float*)d_in[4];
  const float* bo = (const float*)d_in[5];
  float* out = (float*)d_out;

  char* ws = (char*)d_ws;
  const size_t nX = (size_t)NB * SEQ * DMODEL;     // 4,194,304
  const size_t nW = (size_t)DMODEL * DMODEL;       // 1,048,576

  u16* Xb  = (u16*)(ws);                         size_t off = nX * 2;
  u16* Wqb = (u16*)(ws + off);                   off += nW * 2;
  u16* Wkb = (u16*)(ws + off);                   off += nW * 2;
  u16* Wvb = (u16*)(ws + off);                   off += nW * 2;
  u16* Wob = (u16*)(ws + off);                   off += nW * 2;
  u16* Qb  = (u16*)(ws + off);                   off += nX * 2;
  u16* Kb  = (u16*)(ws + off);                   off += nX * 2;
  u16* Vtb = (u16*)(ws + off);                   off += nX * 2;
  u16* Ctx = (u16*)(ws + off);                   off += nX * 2;

  // 1) fp32 -> bf16
  cvt_f32_bf16_kernel<<<(int)((nX + 255) / 256), 256, 0, stream>>>(hs, Xb, (int)nX);
  cvt_f32_bf16_kernel<<<(int)((nW + 255) / 256), 256, 0, stream>>>(Wq, Wqb, (int)nW);
  cvt_f32_bf16_kernel<<<(int)((nW + 255) / 256), 256, 0, stream>>>(Wk, Wkb, (int)nW);
  cvt_f32_bf16_kernel<<<(int)((nW + 255) / 256), 256, 0, stream>>>(Wv, Wvb, (int)nW);
  cvt_f32_bf16_kernel<<<(int)((nW + 255) / 256), 256, 0, stream>>>(Wo, Wob, (int)nW);

  // 2) QKV projections (y selects which weight; V stored transposed)
  qkv_kernel<<<dim3(32 * 16, 3), 256, 0, stream>>>(Xb, Wqb, Wkb, Wvb,
                                                   Qb, Kb, Vtb);

  // 3) causal flash attention (async K/V tile staging)
  attn_kernel<<<(SEQ / 128) * NB * NH, 256, 0, stream>>>(Qb, Kb, Vtb, Ctx);

  // 4) output projection + bias
  out_proj_kernel<<<32 * 16, 256, 0, stream>>>(Ctx, Wob, bo, out);
}
